// MultiHeadAttention_5669356834847
// MI455X (gfx1250) — matrix operand order repaired and checked by reading, not compiled, not run on hardware
//
#include <hip/hip_runtime.h>
#include <hip/hip_bf16.h>
#include <math.h>

// ---------------------------------------------------------------------------
// MultiHeadAttention forward for MI455X (gfx1250, wave32, WMMA).
//   B=2, L=2048, D=512, H=8, DK=DV=64.
// Memory-bound by k_gate read + attn write (~560 MB @ 23.3 TB/s ~= 24 us);
// all GEMMs run on v_wmma_f32_16x16x32_f16 with f32 accumulation. 16x2048
// score rows live in LDS (128 KB/WG -- CDNA5's 320 KB LDS/WGP allows this).
// k_gate is prefetched one tile ahead (global_prefetch); tile staging uses
// async global->LDS loads (ASYNCcnt path) when the builtins are available.
// ---------------------------------------------------------------------------

typedef __attribute__((ext_vector_type(16))) _Float16 v16h;
typedef __attribute__((ext_vector_type(8)))  float    v8f;

#define B_  2
#define L_  2048
#define D_  512
#define H_  8
#define DK_ 64
#define DV_ 64

static constexpr float kInvTemp = 0.125f;   // 1 / sqrt(DK)
static constexpr float kLnEps   = 1e-5f;

#if defined(__has_builtin)
#if __has_builtin(__builtin_amdgcn_global_load_async_to_lds_b128) && \
    __has_builtin(__builtin_amdgcn_s_wait_asynccnt)
#define USE_ASYNC_LDS 1
#endif
#endif

#ifdef USE_ASYNC_LDS
// Builtin takes int4* in global (AS1) / LDS (AS3) address spaces.
typedef int v4i_t __attribute__((ext_vector_type(4)));
typedef __attribute__((address_space(1))) v4i_t* gv4ptr;
typedef __attribute__((address_space(3))) v4i_t* lv4ptr;
#endif

__device__ __forceinline__ v8f vzero8() {
  v8f z = {0.f, 0.f, 0.f, 0.f, 0.f, 0.f, 0.f, 0.f};
  return z;
}

__device__ __forceinline__ v8f wmma_f16(v16h a, v16h b, v8f c) {
  // D = A(16x32) * B(32x16) + C, f32 accumulate
  return __builtin_amdgcn_wmma_f32_16x16x32_f16(
      /*neg_a=*/false, a, /*neg_b=*/false, b,
      /*c_mod=*/(short)0, c, /*reuse_a=*/false, /*reuse_b=*/false);
}

// --- Fragment loaders (layouts per CDNA5 ISA 7.12.2) -----------------------
// A 16x32 f16: lane m = lane%16; half i -> K = 16*(i/8) + 8*(lane/16) + i%8.
__device__ __forceinline__ v16h load_a_frag_f32(const float* base, int ld) {
  const int lane = threadIdx.x & 31;
  const int m = lane & 15, g = lane >> 4;
  const float* row = base + (size_t)m * ld;
  v16h a;
#pragma unroll
  for (int i = 0; i < 8; ++i) a[i] = (_Float16)row[8 * g + i];
#pragma unroll
  for (int i = 0; i < 8; ++i) a[8 + i] = (_Float16)row[16 + 8 * g + i];
  return a;
}

__device__ __forceinline__ v16h load_a_frag_f16(const _Float16* base, int ld) {
  const int lane = threadIdx.x & 31;
  const int m = lane & 15, g = lane >> 4;
  const _Float16* row = base + (size_t)m * ld;
  v16h a;
#pragma unroll
  for (int i = 0; i < 8; ++i) a[i] = row[8 * g + i];
#pragma unroll
  for (int i = 0; i < 8; ++i) a[8 + i] = row[16 + 8 * g + i];
  return a;
}

// A fragment from f32 LDS with a per-row scale (softmax normalization).
__device__ __forceinline__ v16h load_a_frag_f32_scaled(const float* base, int ld,
                                                       const float* rowscale) {
  const int lane = threadIdx.x & 31;
  const int m = lane & 15, g = lane >> 4;
  const float* row = base + (size_t)m * ld;
  const float sc = rowscale[m];
  v16h a;
#pragma unroll
  for (int i = 0; i < 8; ++i) a[i] = (_Float16)(row[8 * g + i] * sc);
#pragma unroll
  for (int i = 0; i < 8; ++i) a[8 + i] = (_Float16)(row[16 + 8 * g + i] * sc);
  return a;
}

// B 32x16 where B[k][n] = W[n][k] (multiply by W^T). W row-major, leading dim ld.
// Lane n = lane%16; half i -> K = 16*(i/8) + 8*(lane/16) + i%8 (same as A).
__device__ __forceinline__ v16h load_bT_frag_f32(const float* W, int ld) {
  const int lane = threadIdx.x & 31;
  const int n = lane & 15, g = lane >> 4;
  const float* row = W + (size_t)n * ld;
  v16h b;
#pragma unroll
  for (int i = 0; i < 8; ++i) b[i] = (_Float16)row[8 * g + i];
#pragma unroll
  for (int i = 0; i < 8; ++i) b[8 + i] = (_Float16)row[16 + 8 * g + i];
  return b;
}

__device__ __forceinline__ v16h load_bT_frag_f16(const _Float16* W, int ld) {
  const int lane = threadIdx.x & 31;
  const int n = lane & 15, g = lane >> 4;
  const _Float16* row = W + (size_t)n * ld;
  v16h b;
#pragma unroll
  for (int i = 0; i < 8; ++i) b[i] = row[8 * g + i];
#pragma unroll
  for (int i = 0; i < 8; ++i) b[8 + i] = row[16 + 8 * g + i];
  return b;
}

// B 32x16 direct from row-major [K][N] source (used for V).
__device__ __forceinline__ v16h load_b_frag_f16(const _Float16* Bp, int ld) {
  const int lane = threadIdx.x & 31;
  const int n = lane & 15, g = lane >> 4;
  v16h b;
#pragma unroll
  for (int i = 0; i < 8; ++i) b[i] = Bp[(size_t)(8 * g + i) * ld + n];
#pragma unroll
  for (int i = 0; i < 8; ++i) b[8 + i] = Bp[(size_t)(16 + 8 * g + i) * ld + n];
  return b;
}

// ---------------------------------------------------------------------------
// Kernel 1: fused Q/K/V projections. out[m,n] = x[m,:] . W[n,:] + bias[n].
// One wave per 16(M) x 64(N) strip; 3 * 256 * 8 = 6144 waves.
// ---------------------------------------------------------------------------
__global__ void proj_kernel(const float* __restrict__ q, const float* __restrict__ k,
                            const float* __restrict__ v,
                            const float* __restrict__ w_qs, const float* __restrict__ b_qs,
                            const float* __restrict__ w_ks, const float* __restrict__ b_ks,
                            const float* __restrict__ w_vs, const float* __restrict__ b_vs,
                            _Float16* __restrict__ Qh, _Float16* __restrict__ Kh,
                            _Float16* __restrict__ Vh) {
  const int wid = (blockIdx.x * blockDim.x + threadIdx.x) >> 5;
  const int mat = wid / (256 * 8);          // 0:Q 1:K 2:V
  const int rem = wid % (256 * 8);
  const int m0 = (rem >> 3) * 16;           // row tile
  const int n0 = (rem & 7) * 64;            // col strip

  const float* src  = (mat == 0) ? q    : (mat == 1) ? k    : v;
  const float* W    = (mat == 0) ? w_qs : (mat == 1) ? w_ks : w_vs;
  const float* bias = (mat == 0) ? b_qs : (mat == 1) ? b_ks : b_vs;
  _Float16*    dst  = (mat == 0) ? Qh   : (mat == 1) ? Kh   : Vh;

  v8f acc[4];
#pragma unroll
  for (int s = 0; s < 4; ++s) acc[s] = vzero8();

  for (int k0 = 0; k0 < D_; k0 += 32) {
    v16h a = load_a_frag_f32(src + (size_t)m0 * D_ + k0, D_);
#pragma unroll
    for (int s = 0; s < 4; ++s) {
      v16h bf = load_bT_frag_f32(W + (size_t)(n0 + s * 16) * D_ + k0, D_);
      acc[s] = wmma_f16(a, bf, acc[s]);
    }
  }

  const int lane = threadIdx.x & 31;
  const int nb = lane & 15, g = lane >> 4;
#pragma unroll
  for (int s = 0; s < 4; ++s) {
    const int n = n0 + s * 16 + nb;
    const float bv = bias[n];
#pragma unroll
    for (int vv = 0; vv < 8; ++vv) {
      const int m = m0 + 8 * g + vv;
      dst[(size_t)m * D_ + n] = (_Float16)(acc[s][vv] + bv);
    }
  }
}

// ---------------------------------------------------------------------------
// Kernel 2: attention. One 128-thread WG per (b, h, 16-row q tile).
// S[16][2048] f32 lives in LDS; 4 waves split the 2048 key columns.
// ---------------------------------------------------------------------------
__global__ void attn_kernel(const _Float16* __restrict__ Qh,
                            const _Float16* __restrict__ Kh,
                            const _Float16* __restrict__ Vh,
                            const unsigned char* __restrict__ mask,  // jnp bool_ (1B)
                            const float* __restrict__ k_gate,
                            float* __restrict__ attn_out,
                            float* __restrict__ O) {
  extern __shared__ char smem[];
  float*    S      = (float*)smem;                                // 16 x 2048 f32
  _Float16* Qt     = (_Float16*)(smem + 16 * L_ * sizeof(float)); // 16 x 64 f16
  float*    red    = (float*)(Qt + 16 * DK_);                     // 16 x 8
  float*    rowmax = red + 16 * 8;                                // 16
  float*    rowinv = rowmax + 16;                                 // 16

  const int blk  = blockIdx.x;            // b*1024 + h*128 + qt
  const int qt   = blk & 127;
  const int h    = (blk >> 7) & (H_ - 1);
  const int b    = blk >> 10;
  const int q0   = qt * 16;
  const int tid  = threadIdx.x;
  const int wave = tid >> 5;
  const int lane = tid & 31;
  const int nb = lane & 15, g = lane >> 4;

  // Stage Q tile (16 x 64 f16 = 2 KB) into LDS.
#ifdef USE_ASYNC_LDS
  {
    const int r  = tid >> 3;          // 16 rows, 8 threads per row
    const int c8 = (tid & 7) * 8;     // 8 halfs = 16 B per thread
    const _Float16* gsrc =
        Qh + (size_t)(b * L_ + q0 + r) * (H_ * DK_) + h * DK_ + c8;
    __builtin_amdgcn_global_load_async_to_lds_b128(
        (gv4ptr)(void*)gsrc, (lv4ptr)(void*)(Qt + r * DK_ + c8), 0, 0);
    __builtin_amdgcn_s_wait_asynccnt(0);
  }
#else
  for (int i = tid; i < 16 * DK_; i += blockDim.x) {
    const int r = i >> 6, c = i & 63;
    Qt[i] = Qh[(size_t)(b * L_ + q0 + r) * (H_ * DK_) + h * DK_ + c];
  }
#endif
  __syncthreads();

  // ---- Phase A: S = (Q K^T) * invTemp * gate, masked -> LDS -------------
  const int kbeg = wave * (L_ / 4);
  const size_t gbase = (size_t)(b * H_ + h) * L_ * L_;
  for (int kt0 = kbeg; kt0 < kbeg + L_ / 4; kt0 += 16) {
    // Prefetch next tile's gate rows (dominant HBM stream): one row per lane.
    if (kt0 + 16 < kbeg + L_ / 4) {
      const int pr = lane & 15;
      __builtin_prefetch(
          &k_gate[gbase + (size_t)(q0 + pr) * L_ + kt0 + 16], 0, 0);
    }
    v8f acc = vzero8();
#pragma unroll
    for (int k0 = 0; k0 < DK_; k0 += 32) {
      v16h a  = load_a_frag_f16(Qt + k0, DK_);
      v16h bf = load_bT_frag_f16(
          Kh + (size_t)(b * L_ + kt0) * (H_ * DK_) + h * DK_ + k0, H_ * DK_);
      acc = wmma_f16(a, bf, acc);
    }
    const int kcol = kt0 + nb;
#pragma unroll
    for (int vv = 0; vv < 8; ++vv) {
      const int m = 8 * g + vv;
      const int qrow = q0 + m;
      float sv = acc[vv] * kInvTemp * k_gate[gbase + (size_t)qrow * L_ + kcol];
      if (mask[((size_t)b * L_ + qrow) * L_ + kcol]) sv = -__builtin_inff();
      S[m * L_ + kcol] = sv;
    }
  }
  __syncthreads();

  // ---- Softmax (8 threads per row) --------------------------------------
  {
    const int r = tid >> 3, sub = tid & 7;
    float mx = -__builtin_inff();
    for (int j = sub; j < L_; j += 8) mx = fmaxf(mx, S[r * L_ + j]);
    red[r * 8 + sub] = mx;
  }
  __syncthreads();
  if ((tid & 7) == 0) {
    const int r = tid >> 3;
    float mx = red[r * 8];
#pragma unroll
    for (int j = 1; j < 8; ++j) mx = fmaxf(mx, red[r * 8 + j]);
    rowmax[r] = mx;
  }
  __syncthreads();
  {
    const int r = tid >> 3, sub = tid & 7;
    const float mx = rowmax[r];
    float sum = 0.f;
    for (int j = sub; j < L_; j += 8) {
      const float e = __expf(S[r * L_ + j] - mx);
      S[r * L_ + j] = e;
      sum += e;
    }
    red[r * 8 + sub] = sum;
  }
  __syncthreads();
  if ((tid & 7) == 0) {
    const int r = tid >> 3;
    float s = 0.f;
#pragma unroll
    for (int j = 0; j < 8; ++j) s += red[r * 8 + j];
    rowinv[r] = 1.0f / s;
  }
  __syncthreads();

  // ---- Normalized attention writeout (coalesced) ------------------------
  {
    const size_t abase = ((size_t)(b * H_ + h) * L_ + q0) * L_;
    for (int i = tid; i < 16 * L_; i += blockDim.x) {
      const int r = i >> 11, c = i & (L_ - 1);
      attn_out[abase + (size_t)r * L_ + c] = S[i] * rowinv[r];
    }
  }

  // ---- Phase C: O = P @ V (wave-split over k, WMMA) ---------------------
  v8f oacc[4];
#pragma unroll
  for (int s = 0; s < 4; ++s) oacc[s] = vzero8();

  for (int k0 = kbeg; k0 < kbeg + L_ / 4; k0 += 32) {
    v16h a = load_a_frag_f32_scaled(S + k0, L_, rowinv);
#pragma unroll
    for (int s = 0; s < 4; ++s) {
      v16h bf = load_b_frag_f16(
          Vh + (size_t)(b * L_ + k0) * (H_ * DV_) + h * DV_ + s * 16, H_ * DV_);
      oacc[s] = wmma_f16(a, bf, oacc[s]);
    }
  }
  __syncthreads();                 // all waves done reading S

  float* Opart = S;                // reuse: 4 x 16 x 64 f32 partials
#pragma unroll
  for (int s = 0; s < 4; ++s) {
#pragma unroll
    for (int vv = 0; vv < 8; ++vv) {
      const int m = 8 * g + vv;
      Opart[(wave * 16 + m) * 64 + s * 16 + nb] = oacc[s][vv];
    }
  }
  __syncthreads();
  for (int i = tid; i < 16 * 64; i += blockDim.x) {
    const int r = i >> 6, c = i & 63;
    const float sum = Opart[r * 64 + c] + Opart[(16 + r) * 64 + c] +
                      Opart[(32 + r) * 64 + c] + Opart[(48 + r) * 64 + c];
    O[(size_t)(b * L_ + q0 + r) * (H_ * DV_) + h * DV_ + c] = sum;
  }
}

// ---------------------------------------------------------------------------
// Kernel 3: out = LayerNorm(O @ w_fc^T + b_fc + residual). 256 threads / 16 rows.
// ---------------------------------------------------------------------------
__global__ void fc_ln_kernel(const float* __restrict__ O,
                             const float* __restrict__ w_fc,
                             const float* __restrict__ b_fc,
                             const float* __restrict__ q_res,
                             const float* __restrict__ ln_g,
                             const float* __restrict__ ln_b,
                             float* __restrict__ out) {
  extern __shared__ char smem[];
  float* X    = (float*)smem;        // 16 x 512 input tile
  float* P    = X + 16 * D_;         // 16 x 512 pre-LN
  float* redS = P + 16 * D_;         // 16 x 16 partial sums
  float* redQ = redS + 256;          // 16 x 16 partial sumsq
  float* mu   = redQ + 256;          // 16
  float* rstd = mu + 16;             // 16

  const int r0   = blockIdx.x * 16;
  const int tid  = threadIdx.x;
  const int wave = tid >> 5;
  const int lane = tid & 31;
  const int nb = lane & 15, g = lane >> 4;

  // Stage O tile (16 x 512 f32 = 32 KB) into LDS.
#ifdef USE_ASYNC_LDS
  for (int i = tid * 4; i < 16 * D_; i += blockDim.x * 4) {
    __builtin_amdgcn_global_load_async_to_lds_b128(
        (gv4ptr)(void*)(O + (size_t)r0 * D_ + i),
        (lv4ptr)(void*)(X + i), 0, 0);
  }
  __builtin_amdgcn_s_wait_asynccnt(0);
#else
  for (int i = tid; i < 16 * D_; i += blockDim.x)
    X[i] = O[(size_t)r0 * D_ + i];
#endif
  __syncthreads();

  const int n0 = wave * 64;
  v8f acc[4];
#pragma unroll
  for (int s = 0; s < 4; ++s) acc[s] = vzero8();

  for (int k0 = 0; k0 < D_; k0 += 32) {
    v16h a = load_a_frag_f32(X + k0, D_);
#pragma unroll
    for (int s = 0; s < 4; ++s) {
      v16h bf = load_bT_frag_f32(w_fc + (size_t)(n0 + s * 16) * D_ + k0, D_);
      acc[s] = wmma_f16(a, bf, acc[s]);
    }
  }

#pragma unroll
  for (int s = 0; s < 4; ++s) {
    const int n = n0 + s * 16 + nb;
    const float bv = b_fc[n];
#pragma unroll
    for (int vv = 0; vv < 8; ++vv) {
      const int m = 8 * g + vv;
      P[m * D_ + n] = acc[s][vv] + bv + q_res[(size_t)(r0 + m) * D_ + n];
    }
  }
  __syncthreads();

  // LayerNorm: 16 threads per row.
  {
    const int r = tid >> 4, sub = tid & 15;
    float s = 0.f, s2 = 0.f;
    for (int c = sub; c < D_; c += 16) {
      const float x = P[r * D_ + c];
      s += x; s2 += x * x;
    }
    redS[r * 16 + sub] = s;
    redQ[r * 16 + sub] = s2;
  }
  __syncthreads();
  if ((tid & 15) == 0) {
    const int r = tid >> 4;
    float s = 0.f, s2 = 0.f;
#pragma unroll
    for (int j = 0; j < 16; ++j) { s += redS[r * 16 + j]; s2 += redQ[r * 16 + j]; }
    const float m = s / (float)D_;
    mu[r] = m;
    rstd[r] = rsqrtf(s2 / (float)D_ - m * m + kLnEps);
  }
  __syncthreads();
  {
    const int r = tid >> 4, sub = tid & 15;
    const float m = mu[r], rv = rstd[r];
    for (int c = sub; c < D_; c += 16)
      out[(size_t)(r0 + r) * D_ + c] = (P[r * D_ + c] - m) * rv * ln_g[c] + ln_b[c];
  }
}

// ---------------------------------------------------------------------------
extern "C" void kernel_launch(void* const* d_in, const int* in_sizes, int n_in,
                              void* d_out, int out_size, void* d_ws, size_t ws_size,
                              hipStream_t stream) {
  (void)in_sizes; (void)n_in; (void)out_size; (void)ws_size;

  const float* q      = (const float*)d_in[0];
  const float* k      = (const float*)d_in[1];
  const float* v      = (const float*)d_in[2];
  const unsigned char* mask = (const unsigned char*)d_in[3];  // jnp bool_ = 1 byte
  const float* k_gate = (const float*)d_in[4];
  const float* w_qs   = (const float*)d_in[5];
  const float* b_qs   = (const float*)d_in[6];
  const float* w_ks   = (const float*)d_in[7];
  const float* b_ks   = (const float*)d_in[8];
  const float* w_vs   = (const float*)d_in[9];
  const float* b_vs   = (const float*)d_in[10];
  const float* w_fc   = (const float*)d_in[11];
  const float* b_fc   = (const float*)d_in[12];
  const float* ln_g   = (const float*)d_in[13];
  const float* ln_b   = (const float*)d_in[14];

  float* out  = (float*)d_out;                         // [B, L, D]
  float* attn = out + (size_t)B_ * L_ * D_;            // [B, H, L, L]

  // Workspace carve-up: Qh/Kh/Vh (f16) + O (f32)  ~20 MB total.
  char* ws = (char*)d_ws;
  const size_t projBytes = (size_t)B_ * L_ * H_ * DK_ * sizeof(_Float16);
  _Float16* Qh = (_Float16*)ws;               ws += projBytes;
  _Float16* Kh = (_Float16*)ws;               ws += projBytes;
  _Float16* Vh = (_Float16*)ws;               ws += projBytes;
  float*    O  = (float*)ws;

  // 1) Projections: 6144 waves -> 768 blocks x 256 threads.
  proj_kernel<<<768, 256, 0, stream>>>(q, k, v, w_qs, b_qs, w_ks, b_ks,
                                       w_vs, b_vs, Qh, Kh, Vh);

  // 2) Attention: one WG per (b,h,q-tile). Dynamic LDS ~131 KB.
  const size_t smem2 = (size_t)16 * L_ * sizeof(float)       // S
                     + (size_t)16 * DK_ * sizeof(_Float16)   // Qt
                     + (16 * 8 + 16 + 16) * sizeof(float);   // red + rowmax + rowinv
  attn_kernel<<<B_ * H_ * (L_ / 16), 128, smem2, stream>>>(
      Qh, Kh, Vh, mask, k_gate, attn, O);

  // 3) FC + residual + LayerNorm.
  const size_t smem3 = (size_t)(16 * D_ * 2 + 256 * 2 + 32) * sizeof(float);
  fc_ln_kernel<<<(B_ * L_) / 16, 256, smem3, stream>>>(
      O, w_fc, b_fc, q, ln_g, ln_b, out);
}
